// TimedOrthogonalBivectorBlock_429496730089
// MI455X (gfx1250) — compile-verified, matrix operand order checked by reading, not run on hardware
//
#include <hip/hip_runtime.h>
#include <hip/hip_bf16.h>
#include <math.h>

// ---------------------------------------------------------------------------
// Phasor linear-attention block for MI455X (gfx1250, wave32, WMMA bf16 + TDM).
//
//   S[l,t] = sum_p cos(qph_p[l] - kph_p[t]) = phi_q[l].phi_k[t]  (content)
//   plus positional phasor features; set-softmax / sigmoid(pos_weight) folded
//   into phi_q.  96 padded features (64 content | 16 pos | 16 zero).
//   Chunked causal linear attention (chunk=64): intra-chunk masked S@V via
//   WMMA + cross-chunk phi_q @ M with M = exclusive prefix of U_c = V_c^T K_c.
// ---------------------------------------------------------------------------

#define NTOK   4096
#define SEQL   2048
#define DMODEL 512
#define NPROJ  1088          // 32 kphase | 32 qphase | 512 gate | 512 V
#define FPAD   96
#define CHUNK  64
#define NCHUNK 64
#define CPB    32

typedef __attribute__((ext_vector_type(16))) __bf16 bf16x16;
typedef __attribute__((ext_vector_type(8)))  __bf16 bf16x8;
typedef __attribute__((ext_vector_type(8)))  float  f32x8;
typedef __attribute__((ext_vector_type(4)))  unsigned int u32x4;
typedef __attribute__((ext_vector_type(8)))  int i32x8;
typedef __attribute__((ext_vector_type(4)))  int i32x4;

__device__ __forceinline__ f32x8 wmma_bf16(bf16x16 a, bf16x16 b, f32x8 c) {
  return __builtin_amdgcn_wmma_f32_16x16x32_bf16(false, a, false, b,
                                                 (short)0, c, false, false);
}

// A/B fragment loader (ISA 16-bit layout: lane<16 row=lane K={0..7,16..23},
// lane>=16 same row K={8..15,24..31}; B mirrored with lane = column).
__device__ __forceinline__ bf16x16 load_frag(const __bf16* base, int stride) {
  const int lane = threadIdx.x & 31;
  const int row  = lane & 15;
  const int half = lane >> 4;
  const __bf16* p = base + (size_t)row * stride + half * 8;
  bf16x8 a = *(const bf16x8*)(p);
  bf16x8 b = *(const bf16x8*)(p + 16);
  return __builtin_shufflevector(a, b, 0,1,2,3,4,5,6,7,8,9,10,11,12,13,14,15);
}

// ---------------------------------------------------------------------------
// Tensor Data Mover: 2-D bf16 tile load into LDS (D# per ISA 8.3/8.4).
// ---------------------------------------------------------------------------
#if __has_builtin(__builtin_amdgcn_tensor_load_to_lds)
#define HAVE_TDM 1
typedef __attribute__((address_space(3))) const void lds_cv;
__device__ __forceinline__ unsigned int lds_off(const void* p) {
  return (unsigned int)(unsigned long long)(lds_cv*)p;
}
__device__ __forceinline__ void tdm_load_2d_bf16(unsigned int ldsaddr,
                                                 unsigned long long gaddr,
                                                 unsigned int tensor_w,
                                                 unsigned int tensor_h,
                                                 unsigned int tile_w,
                                                 unsigned int tile_h,
                                                 unsigned int row_stride) {
  u32x4 g0;
  g0.x = 1u;                                            // count=1 (valid D#)
  g0.y = ldsaddr;                                       // lds_addr
  g0.z = (unsigned int)(gaddr & 0xffffffffull);         // global_addr lo
  g0.w = (unsigned int)((gaddr >> 32) & 0x1ffffffull)   // global_addr hi
       | (2u << 30);                                    // type = 2 (image)
  i32x8 g1;
  g1[0] = (int)(1u << 16);                              // data_size = 2 bytes
  g1[1] = (int)(tensor_w << 16);                        // tensor_dim0[15:0]
  g1[2] = (int)((tensor_w >> 16) | (tensor_h << 16));   // dim0 hi | dim1 lo
  g1[3] = (int)((tensor_h >> 16) | (tile_w << 16));     // dim1 hi | tile_dim0
  g1[4] = (int)(tile_h & 0xffffu);                      // tile_dim1 (dim2=0)
  g1[5] = (int)row_stride;                              // dim0_stride lo32
  g1[6] = 0;                                            // stride hi / d1s lo
  g1[7] = 0;
  i32x4 z4 = {0, 0, 0, 0};
#if defined(__clang_major__) && __clang_major__ >= 23
  i32x8 z8 = {0, 0, 0, 0, 0, 0, 0, 0};
  __builtin_amdgcn_tensor_load_to_lds(g0, g1, z4, z4, z8, 0);
#else
  __builtin_amdgcn_tensor_load_to_lds(g0, g1, z4, z4, 0);
#endif
}
#endif

// ---------------------------------------------------------------------------
// Kernel 1: Y(4096x1088) = x @ [Wk|Wq|Wg|Wv] + bias
// ---------------------------------------------------------------------------
__global__ __launch_bounds__(256)
void k_proj_gemm(const float* __restrict__ x,
                 const float* __restrict__ Wk, const float* __restrict__ bk,
                 const float* __restrict__ Wq, const float* __restrict__ bq,
                 const float* __restrict__ Wg, const float* __restrict__ bg,
                 const float* __restrict__ Wv, const float* __restrict__ bv,
                 float* __restrict__ Y) {
  __shared__ __bf16 As[128 * 32];
  __shared__ __bf16 Bs[64 * 32];
  const int tid  = threadIdx.x;
  const int wave = tid >> 5;
  const int lane = tid & 31;
  const int m0 = blockIdx.x * 128;
  const int n0 = blockIdx.y * 64;
  const int awb = (wave & 3) * 32;
  const int bwb = (wave >> 2) * 32;
  f32x8 acc[2][2] = {};

  // hoisted per-thread staging coordinates + weight source (column is fixed)
  const int mm = tid >> 3, kqa = (tid & 7) * 4;
  const int nb = tid >> 2, kqb = (tid & 3) * 8;
  const int c = n0 + nb;
  const float* wsrc; int wstride;
  if      (c < 32)  { wsrc = Wk + c;         wstride = 32;  }
  else if (c < 64)  { wsrc = Wq + (c - 32);  wstride = 32;  }
  else if (c < 576) { wsrc = Wg + (c - 64);  wstride = 512; }
  else              { wsrc = Wv + (c - 576); wstride = 512; }
  const float* xrow = x + (size_t)(m0 + mm) * DMODEL;

  for (int kk = 0; kk < 16; ++kk) {
    const int kb = kk * 32;
    #pragma unroll
    for (int r = 0; r < 4; ++r) {           // A: 128x32 fp32 -> bf16
      const float4 v = *(const float4*)(xrow + (size_t)r * 32 * DMODEL + kb + kqa);
      __bf16* dst = &As[(mm + r * 32) * 32 + kqa];
      dst[0] = (__bf16)v.x; dst[1] = (__bf16)v.y;
      dst[2] = (__bf16)v.z; dst[3] = (__bf16)v.w;
    }
    #pragma unroll
    for (int j = 0; j < 8; ++j)             // B: weight tile -> [n][k] bf16
      Bs[nb * 32 + kqb + j] = (__bf16)wsrc[(size_t)(kb + kqb + j) * wstride];
    if (kk + 1 < 16)
      __builtin_prefetch(xrow + kb + 32, 0, 0);
    __syncthreads();
    bf16x16 a0 = load_frag(&As[(awb +  0) * 32], 32);
    bf16x16 a1 = load_frag(&As[(awb + 16) * 32], 32);
    bf16x16 b0 = load_frag(&Bs[(bwb +  0) * 32], 32);
    bf16x16 b1 = load_frag(&Bs[(bwb + 16) * 32], 32);
    acc[0][0] = wmma_bf16(a0, b0, acc[0][0]);
    acc[0][1] = wmma_bf16(a0, b1, acc[0][1]);
    acc[1][0] = wmma_bf16(a1, b0, acc[1][0]);
    acc[1][1] = wmma_bf16(a1, b1, acc[1][1]);
    __syncthreads();
  }
  #pragma unroll
  for (int i = 0; i < 2; ++i)
    #pragma unroll
    for (int j = 0; j < 2; ++j) {
      int col = n0 + bwb + j * 16 + (lane & 15);
      float bias;
      if      (col < 32)  bias = bk[col];
      else if (col < 64)  bias = bq[col - 32];
      else if (col < 576) bias = bg[col - 64];
      else                bias = bv[col - 576];
      int mrow = m0 + awb + i * 16 + (lane >> 4) * 8;
      #pragma unroll
      for (int r = 0; r < 8; ++r)
        Y[(size_t)(mrow + r) * NPROJ + col] = acc[i][j][r] + bias;
    }
}

// ---------------------------------------------------------------------------
// Kernel 2: phasor features, one wave per token (lane-coalesced phase reads)
// ---------------------------------------------------------------------------
__global__ __launch_bounds__(256)
void k_features(const float* __restrict__ Y,
                const float* __restrict__ set_weights,
                const float* __restrict__ pos_freqs,
                const float* __restrict__ pos_weight,
                __bf16* __restrict__ phiQ, __bf16* __restrict__ phiK) {
  const int lane = threadIdx.x & 31;
  const int i = blockIdx.x * 8 + (threadIdx.x >> 5);   // token
  const int l = i & (SEQL - 1);
  const float PI_ = 3.14159265358979323846f;
  float sw[4]; float mx = -1e30f;
  #pragma unroll
  for (int s = 0; s < 4; ++s) { sw[s] = set_weights[s]; mx = fmaxf(mx, sw[s]); }
  float den = 0.f;
  #pragma unroll
  for (int s = 0; s < 4; ++s) { sw[s] = expf(sw[s] - mx); den += sw[s]; }
  const float* y = Y + (size_t)i * NPROJ;
  {                                         // content plane j = lane (0..31)
    const float w = sw[lane >> 3] / den;
    float kph = tanhf(y[lane])      * PI_;
    float qph = tanhf(y[32 + lane]) * PI_;
    int f = 2 * lane;
    phiK[(size_t)i * FPAD + f]     = (__bf16)cosf(kph);
    phiK[(size_t)i * FPAD + f + 1] = (__bf16)sinf(kph);
    phiQ[(size_t)i * FPAD + f]     = (__bf16)(w * cosf(qph));
    phiQ[(size_t)i * FPAD + f + 1] = (__bf16)(w * sinf(qph));
  }
  if (lane < 8) {                           // positional plane p = lane
    float pwv = 1.f / (1.f + expf(-pos_weight[0]));
    float th = 6.283185307179586f * pos_freqs[lane] * (float)l;
    int f = 64 + 2 * lane;
    phiK[(size_t)i * FPAD + f]     = (__bf16)cosf(th);
    phiK[(size_t)i * FPAD + f + 1] = (__bf16)sinf(th);
    phiQ[(size_t)i * FPAD + f]     = (__bf16)(pwv * cosf(th));
    phiQ[(size_t)i * FPAD + f + 1] = (__bf16)(pwv * sinf(th));
  }
  if (lane < 16) {                          // zero pad f = 80..95
    phiK[(size_t)i * FPAD + 80 + lane] = (__bf16)0.f;
    phiQ[(size_t)i * FPAD + 80 + lane] = (__bf16)0.f;
  }
}

// ---------------------------------------------------------------------------
// Kernel 3: gate = sigmoid(Y[:,64:576])  (fully coalesced)
// ---------------------------------------------------------------------------
__global__ __launch_bounds__(256)
void k_gate(const float* __restrict__ Y, float* __restrict__ gate) {
  size_t idx = (size_t)blockIdx.x * 256 + threadIdx.x;   // < NTOK*DMODEL
  int i = (int)(idx >> 9), d = (int)(idx & 511);
  gate[idx] = 1.f / (1.f + expf(-Y[(size_t)i * NPROJ + 64 + d]));
}

// ---------------------------------------------------------------------------
// Kernel 4: VT(512 x 4096 bf16) = transpose(Y[:,576:1088]) via LDS tiles
// ---------------------------------------------------------------------------
__global__ __launch_bounds__(256)
void k_transpose_v(const float* __restrict__ Y, __bf16* __restrict__ VT) {
  __shared__ __bf16 T[64 * 65];             // 65: bank-conflict pad
  const int tok0 = blockIdx.x * 64;
  const int d0   = blockIdx.y * 64;
  #pragma unroll
  for (int e = 0; e < 16; ++e) {
    int idx = threadIdx.x + e * 256;        // < 4096
    int t = idx >> 6, cc = idx & 63;
    T[t * 65 + cc] = (__bf16)Y[(size_t)(tok0 + t) * NPROJ + 576 + d0 + cc];
  }
  __syncthreads();
  #pragma unroll
  for (int e = 0; e < 16; ++e) {
    int idx = threadIdx.x + e * 256;
    int d = idx >> 6, t = idx & 63;
    VT[(size_t)(d0 + d) * NTOK + tok0 + t] = T[t * 65 + d];
  }
}

// ---------------------------------------------------------------------------
// Kernel 5: phiKT(96 x 4096) = transpose(phiK) via LDS tiles
// ---------------------------------------------------------------------------
__global__ __launch_bounds__(256)
void k_transpose_phiK(const __bf16* __restrict__ phiK,
                      __bf16* __restrict__ phiKT) {
  __shared__ __bf16 T[64 * FPAD];
  const int tok0 = blockIdx.x * 64;
  #pragma unroll
  for (int e = 0; e < 24; ++e) {
    int idx = threadIdx.x + e * 256;        // < 6144
    int t = idx / FPAD, f = idx - t * FPAD;
    T[t * FPAD + f] = phiK[(size_t)(tok0 + t) * FPAD + f];
  }
  __syncthreads();
  #pragma unroll
  for (int e = 0; e < 24; ++e) {
    int idx = threadIdx.x + e * 256;
    int f = idx >> 6, t = idx & 63;
    phiKT[(size_t)f * NTOK + tok0 + t] = T[t * FPAD + f];
  }
}

// ---------------------------------------------------------------------------
// Kernel 6: per-chunk state U_c(512x96) = V_c^T @ phiK_c  (WMMA)
// ---------------------------------------------------------------------------
__global__ __launch_bounds__(256)
void k_chunk_state(const __bf16* __restrict__ VT,
                   const __bf16* __restrict__ phiKT,
                   float* __restrict__ U) {
  const int chunk = blockIdx.x;
  const int tok0  = chunk * CHUNK;
  const int wave  = threadIdx.x >> 5;
  const int lane  = threadIdx.x & 31;
  float* u = U + (size_t)chunk * (DMODEL * FPAD);
  for (int mt = 0; mt < 4; ++mt) {
    const int d0 = (wave * 4 + mt) * 16;
    f32x8 acc[6] = {};
    #pragma unroll
    for (int kk = 0; kk < 2; ++kk) {
      bf16x16 a = load_frag(VT + (size_t)d0 * NTOK + tok0 + kk * 32, NTOK);
      #pragma unroll
      for (int nt = 0; nt < 6; ++nt) {
        bf16x16 b = load_frag(phiKT + (size_t)(nt * 16) * NTOK + tok0 + kk * 32, NTOK);
        acc[nt] = wmma_bf16(a, b, acc[nt]);
      }
    }
    #pragma unroll
    for (int nt = 0; nt < 6; ++nt) {
      int f  = nt * 16 + (lane & 15);
      int dr = d0 + (lane >> 4) * 8;
      #pragma unroll
      for (int r = 0; r < 8; ++r)
        u[(size_t)(dr + r) * FPAD + f] = acc[nt][r];
    }
  }
}

// ---------------------------------------------------------------------------
// Kernel 7: exclusive prefix of U over chunks -> Mt (bf16)
// ---------------------------------------------------------------------------
__global__ __launch_bounds__(256)
void k_prefix(const float* __restrict__ U, __bf16* __restrict__ Mt) {
  int idx = blockIdx.x * 256 + threadIdx.x;
  if (idx >= 2 * DMODEL * FPAD) return;
  int b   = idx / (DMODEL * FPAD);
  int rem = idx - b * (DMODEL * FPAD);
  float acc = 0.f;
  for (int cb = 0; cb < CPB; ++cb) {
    size_t o = (size_t)(b * CPB + cb) * (DMODEL * FPAD) + rem;
    Mt[o] = (__bf16)acc;
    acc  += U[o];
  }
}

// ---------------------------------------------------------------------------
// Kernel 8: chunked causal attention + gating (2 chunks / block, per-wave
// LDS score-strip transpose, wave-local s_wait_dscnt only)
// ---------------------------------------------------------------------------
__global__ __launch_bounds__(256)
void k_attn(const __bf16* __restrict__ phiQ, const __bf16* __restrict__ phiK,
            const __bf16* __restrict__ VT,  const __bf16* __restrict__ Mt,
            const float* __restrict__ gate, __bf16* __restrict__ Tn) {
  __shared__ __bf16 stC[8][16 * CHUNK];
  __shared__ __bf16 stP[8][16 * CHUNK];
  const int wave  = threadIdx.x >> 5;
  const int lane  = threadIdx.x & 31;
  const int chunk = blockIdx.x * 2 + (wave >> 2);
  const int row0  = (wave & 3) * 16;
  const int tok0  = chunk * CHUNK;

  bf16x16 qf[3];
  #pragma unroll
  for (int kf = 0; kf < 3; ++kf)
    qf[kf] = load_frag(phiQ + (size_t)(tok0 + row0) * FPAD + kf * 32, FPAD);

  for (int st = 0; st < 4; ++st) {
    const int tb    = st * 16;
    const int tloc  = tb + (lane & 15);
    const int mbase = (lane >> 4) * 8;
    if (tb > row0 + 15) {
      #pragma unroll
      for (int r = 0; r < 8; ++r) {
        stC[wave][(mbase + r) * CHUNK + tloc] = (__bf16)0.f;
        stP[wave][(mbase + r) * CHUNK + tloc] = (__bf16)0.f;
      }
      continue;
    }
    const __bf16* pk = phiK + (size_t)(tok0 + tb) * FPAD;
    bf16x16 b0 = load_frag(pk +  0, FPAD);
    bf16x16 b1 = load_frag(pk + 32, FPAD);
    bf16x16 b2 = load_frag(pk + 64, FPAD);
    f32x8 sc = {}, sp = {};
    sc = wmma_bf16(qf[0], b0, sc);
    sc = wmma_bf16(qf[1], b1, sc);
    sp = wmma_bf16(qf[2], b2, sp);
    #pragma unroll
    for (int r = 0; r < 8; ++r) {
      int m = mbase + r;
      bool keep = (tloc <= row0 + m);
      stC[wave][m * CHUNK + tloc] = keep ? (__bf16)sc[r] : (__bf16)0.f;
      stP[wave][m * CHUNK + tloc] = keep ? (__bf16)sp[r] : (__bf16)0.f;
    }
  }
  asm volatile("s_wait_dscnt 0x0" ::: "memory");

  const int ktmax = (row0 + 15) >> 5;

  for (int seg = 0; seg < 8; ++seg) {
    f32x8 accC[4] = {}, accP[4] = {};
    for (int kt = 0; kt <= ktmax; ++kt) {
      bf16x16 ac = load_frag(&stC[wave][kt * 32], CHUNK);
      bf16x16 ap = load_frag(&stP[wave][kt * 32], CHUNK);
      #pragma unroll
      for (int dt = 0; dt < 4; ++dt) {
        int d0 = seg * 64 + dt * 16;
        bf16x16 bv = load_frag(VT + (size_t)d0 * NTOK + tok0 + kt * 32, NTOK);
        accC[dt] = wmma_bf16(ac, bv, accC[dt]);
        accP[dt] = wmma_bf16(ap, bv, accP[dt]);
      }
    }
    const __bf16* mt = Mt + (size_t)chunk * (DMODEL * FPAD);
    #pragma unroll
    for (int dt = 0; dt < 4; ++dt) {
      int d0 = seg * 64 + dt * 16;
      bf16x16 m0 = load_frag(mt + (size_t)d0 * FPAD +  0, FPAD);
      bf16x16 m1 = load_frag(mt + (size_t)d0 * FPAD + 32, FPAD);
      bf16x16 m2 = load_frag(mt + (size_t)d0 * FPAD + 64, FPAD);
      accC[dt] = wmma_bf16(qf[0], m0, accC[dt]);
      accC[dt] = wmma_bf16(qf[1], m1, accC[dt]);
      accP[dt] = wmma_bf16(qf[2], m2, accP[dt]);
    }
    #pragma unroll
    for (int dt = 0; dt < 4; ++dt) {
      int d    = seg * 64 + dt * 16 + (lane & 15);
      int mrow = row0 + (lane >> 4) * 8;
      #pragma unroll
      for (int r = 0; r < 8; ++r) {
        int lc = mrow + r;
        int i  = tok0 + lc;
        int lb = i & (SEQL - 1);
        float g = gate[(size_t)i * DMODEL + d];
        float t = g * accC[dt][r] + (1.f - g) * accP[dt][r];
        t *= rsqrtf((float)(lb + 1) * 8.0f);
        Tn[(size_t)i * DMODEL + d] = (__bf16)t;
      }
    }
  }
}

// ---------------------------------------------------------------------------
// Kernel 9: out = x + T @ Wo + bo  (A tile staged by the Tensor Data Mover)
// ---------------------------------------------------------------------------
__global__ __launch_bounds__(256)
void k_out_gemm(const __bf16* __restrict__ Tn, const float* __restrict__ Wo,
                const float* __restrict__ bo, const float* __restrict__ x,
                float* __restrict__ out) {
  __shared__ __bf16 As[128 * 32];
  __shared__ __bf16 Bs[64 * 32];
  const int tid  = threadIdx.x;
  const int wave = tid >> 5;
  const int lane = tid & 31;
  const int m0 = blockIdx.x * 128;
  const int n0 = blockIdx.y * 64;
  const int awb = (wave & 3) * 32;
  const int bwb = (wave >> 2) * 32;
  f32x8 acc[2][2] = {};
  const int nb = tid >> 2, kqb = (tid & 3) * 8;
  const float* wcol = Wo + n0 + nb;

  for (int kk = 0; kk < 16; ++kk) {
    const int kb = kk * 32;
#if defined(HAVE_TDM)
    if (wave == 0)   // one DMA issue; TDM ignores EXEC, tracked by TENSORcnt
      tdm_load_2d_bf16(lds_off(As),
                       (unsigned long long)(uintptr_t)(Tn + (size_t)m0 * DMODEL + kb),
                       /*tensor_w=*/DMODEL, /*tensor_h=*/NTOK - m0,
                       /*tile_w=*/32, /*tile_h=*/128, /*row_stride=*/DMODEL);
#else
    {
      int m = tid >> 1, kq = (tid & 1) * 16;
      const uint4* src = (const uint4*)(Tn + (size_t)(m0 + m) * DMODEL + kb + kq);
      *(uint4*)&As[m * 32 + kq]     = src[0];
      *(uint4*)&As[m * 32 + kq + 8] = src[1];
    }
#endif
    #pragma unroll
    for (int j = 0; j < 8; ++j)             // B: Wo fp32 [k][n] -> bf16 [n][k]
      Bs[nb * 32 + kqb + j] = (__bf16)wcol[(size_t)(kb + kqb + j) * DMODEL];
    if (kk + 1 < 16)
      __builtin_prefetch(wcol + (size_t)(kb + 32) * DMODEL, 0, 0);
#if defined(HAVE_TDM)
    if (wave == 0)
      __builtin_amdgcn_s_wait_tensorcnt(0);
#endif
    __syncthreads();
    bf16x16 a0 = load_frag(&As[(awb +  0) * 32], 32);
    bf16x16 a1 = load_frag(&As[(awb + 16) * 32], 32);
    bf16x16 b0 = load_frag(&Bs[(bwb +  0) * 32], 32);
    bf16x16 b1 = load_frag(&Bs[(bwb + 16) * 32], 32);
    acc[0][0] = wmma_bf16(a0, b0, acc[0][0]);
    acc[0][1] = wmma_bf16(a0, b1, acc[0][1]);
    acc[1][0] = wmma_bf16(a1, b0, acc[1][0]);
    acc[1][1] = wmma_bf16(a1, b1, acc[1][1]);
    __syncthreads();
  }
  #pragma unroll
  for (int i = 0; i < 2; ++i)
    #pragma unroll
    for (int j = 0; j < 2; ++j) {
      int col  = n0 + bwb + j * 16 + (lane & 15);
      int mrow = m0 + awb + i * 16 + (lane >> 4) * 8;
      float bias = bo[col];
      #pragma unroll
      for (int r = 0; r < 8; ++r) {
        size_t o = (size_t)(mrow + r) * DMODEL + col;
        out[o] = x[o] + acc[i][j][r] + bias;
      }
    }
}

// ---------------------------------------------------------------------------
extern "C" void kernel_launch(void* const* d_in, const int* in_sizes, int n_in,
                              void* d_out, int out_size, void* d_ws, size_t ws_size,
                              hipStream_t stream) {
  const float* x  = (const float*)d_in[0];
  const float* Wk = (const float*)d_in[1];
  const float* bk = (const float*)d_in[2];
  const float* Wq = (const float*)d_in[3];
  const float* bq = (const float*)d_in[4];
  const float* Wv = (const float*)d_in[5];
  const float* bv = (const float*)d_in[6];
  const float* sw = (const float*)d_in[7];
  const float* pf = (const float*)d_in[8];
  const float* Wg = (const float*)d_in[9];
  const float* bg = (const float*)d_in[10];
  const float* pw = (const float*)d_in[11];
  const float* Wo = (const float*)d_in[12];
  const float* bo = (const float*)d_in[13];

  char* ws = (char*)d_ws;
  size_t off = 0;
  auto take = [&](size_t bytes) {
    size_t o = off;
    off = (off + bytes + 255) & ~(size_t)255;
    return o;
  };
  float*  Y     = (float*) (ws + take(sizeof(float)  * NTOK * NPROJ));
  __bf16* phiQ  = (__bf16*)(ws + take(sizeof(__bf16) * NTOK * FPAD));
  __bf16* phiK  = (__bf16*)(ws + take(sizeof(__bf16) * NTOK * FPAD));
  __bf16* phiKT = (__bf16*)(ws + take(sizeof(__bf16) * FPAD * NTOK));
  __bf16* VT    = (__bf16*)(ws + take(sizeof(__bf16) * DMODEL * NTOK));
  float*  gatep = (float*) (ws + take(sizeof(float)  * NTOK * DMODEL));
  float*  U     = (float*) (ws + take(sizeof(float)  * NCHUNK * DMODEL * FPAD));
  __bf16* Mt    = (__bf16*)(ws + take(sizeof(__bf16) * NCHUNK * DMODEL * FPAD));
  __bf16* Tnb   = (__bf16*)(ws + take(sizeof(__bf16) * NTOK * DMODEL));

  k_proj_gemm    <<<dim3(NTOK / 128, NPROJ / 64), 256, 0, stream>>>(
      x, Wk, bk, Wq, bq, Wg, bg, Wv, bv, Y);
  k_features     <<<NTOK / 8, 256, 0, stream>>>(Y, sw, pf, pw, phiQ, phiK);
  k_gate         <<<(NTOK * DMODEL) / 256, 256, 0, stream>>>(Y, gatep);
  k_transpose_v  <<<dim3(NTOK / 64, DMODEL / 64), 256, 0, stream>>>(Y, VT);
  k_transpose_phiK<<<NTOK / 64, 256, 0, stream>>>(phiK, phiKT);
  k_chunk_state  <<<NCHUNK, 256, 0, stream>>>(VT, phiKT, U);
  k_prefix       <<<(2 * DMODEL * FPAD + 255) / 256, 256, 0, stream>>>(U, Mt);
  k_attn         <<<NCHUNK / 2, 256, 0, stream>>>(phiQ, phiK, VT, Mt, gatep, Tnb);
  k_out_gemm     <<<dim3(NTOK / 128, DMODEL / 64), 256, 0, stream>>>(
      Tnb, Wo, bo, x, (float*)d_out);
}
